// DysonBlock_24060406792902
// MI455X (gfx1250) — compile-verified, matrix operand order checked by reading, not run on hardware
//
#include <hip/hip_runtime.h>
#include <hip/hip_bf16.h>

// DysonBlock on MI455X (gfx1250, wave32):
//   1) depthwise circular conv(K=4) + SiLU on g          -> gact
//   2) u = (x @ w_in) * gact    : WMMA f16 hi/lo GEMM (fp32-accurate split)
//   3) diagonal SSM recurrence (exact equivalent of the FFT conv):
//        s_n[t] = lambda_n s_n[t-1] + u[t],  y[t] = sum_n w_n s_n[t]
//      one wave32 per (b,h), 2 complex modes per lane, shfl_xor reduce,
//      u streamed through LDS with async-to-LDS double buffering
//   4) out = y @ w_out          : same WMMA GEMM
//
// Precision: fp32 values split into f16 hi + f16 lo; each K-step does
//   C += A_hi*B_hi + A_hi*B_lo + A_lo*B_hi   (3x v_wmma_f32_16x16x32_f16)

typedef __attribute__((ext_vector_type(16))) _Float16 v16h;
typedef __attribute__((ext_vector_type(8)))  _Float16 v8h;
typedef __attribute__((ext_vector_type(8)))  float    v8f;

static constexpr int Bn = 8, Ln = 2048, Dn = 1024, Nmod = 64, Hn = 512, KSn = 4;

// ---------------------------------------------------------------- conv+silu
__global__ void dyson_conv_silu(const float* __restrict__ g,
                                const float* __restrict__ ker,   // (KS,1,D)
                                const float* __restrict__ bias,  // (D)
                                float* __restrict__ gact) {
  size_t i = (size_t)blockIdx.x * blockDim.x + threadIdx.x;
  size_t total = (size_t)Bn * Ln * Dn;
  if (i >= total) return;
  int d = (int)(i % Dn);
  size_t bt = i / Dn;
  int t = (int)(bt % Ln);
  int b = (int)(bt / Ln);
  float acc = bias[d];
#pragma unroll
  for (int k = 0; k < KSn; ++k) {
    int tt = t + k - 1;              // pad_before = (KS-1)/2 = 1, circular
    tt += (tt < 0) ? Ln : 0;
    tt -= (tt >= Ln) ? Ln : 0;
    acc += g[((size_t)b * Ln + tt) * Dn + d] * ker[k * Dn + d];
  }
  gact[i] = acc / (1.0f + expf(-acc));   // silu
}

// ------------------------------------------------- fp32 -> f16 hi/lo split
__global__ void dyson_split(const float* __restrict__ in,
                            _Float16* __restrict__ hi,
                            _Float16* __restrict__ lo, size_t n) {
  size_t i = (size_t)blockIdx.x * blockDim.x + threadIdx.x;
  if (i >= n) return;
  float v = in[i];
  _Float16 h = (_Float16)v;
  hi[i] = h;
  lo[i] = (_Float16)(v - (float)h);
}

// --------------------------- weight transpose (to [N][K]) + hi/lo split
__global__ void dyson_transpose_split(const float* __restrict__ w,  // [K][N]
                                      _Float16* __restrict__ hi,    // [N][K]
                                      _Float16* __restrict__ lo,
                                      int K, int N) {
  size_t i = (size_t)blockIdx.x * blockDim.x + threadIdx.x;
  if (i >= (size_t)K * N) return;
  int n = (int)(i / K);
  int k = (int)(i % K);
  float v = w[(size_t)k * N + n];
  _Float16 h = (_Float16)v;
  hi[i] = h;
  lo[i] = (_Float16)(v - (float)h);
}

// ------------------------------------------- lambda_n = exp(A_n) per (h,n)
__global__ void dyson_lambda(const float* __restrict__ log_decay,
                             const float* __restrict__ freq,
                             float* __restrict__ lamr,
                             float* __restrict__ lami, int n) {
  int i = blockIdx.x * blockDim.x + threadIdx.x;
  if (i >= n) return;
  float r = expf(-expf(log_decay[i]));   // |lambda| = exp(-exp(ld)) < 1
  float f = freq[i];
  lamr[i] = r * cosf(f);
  lami[i] = r * sinf(f);
}

// ------------------------------------------------------- WMMA hi/lo GEMM
// A: [M][K] f16 hi/lo (row major).  B: [N][K] f16 hi/lo (pre-transposed).
// Wave computes a 32(M) x 64(N) block: B fragments loaded once, used by
// both 16-row A tiles. Fragment layout per ISA 16-bit A 16x32:
//   lanes 0-15 row M=l hold K = {kb..kb+7, kb+16..kb+23}
//   lanes 16-31 row M=l hold K = {kb+8..kb+15, kb+24..kb+31}
__device__ __forceinline__ v16h dyson_frag(const _Float16* p) {
  v8h c0 = *(const v8h*)(p);        // 16B chunk, K offsets +0..7 (per half)
  v8h c1 = *(const v8h*)(p + 16);   // 16B chunk, K offsets +16..23
  v16h r;
#pragma unroll
  for (int i = 0; i < 8; ++i) { r[i] = c0[i]; r[i + 8] = c1[i]; }
  return r;
}

__global__ void dyson_gemm_hilo(const _Float16* __restrict__ Ahi,
                                const _Float16* __restrict__ Alo,
                                const _Float16* __restrict__ Bhi,
                                const _Float16* __restrict__ Blo,
                                const float* gmul,   // optional, may alias out
                                float* out,          // [M][N] fp32
                                int M, int N, int K) {
  int wave = (int)(((size_t)blockIdx.x * blockDim.x + threadIdx.x) >> 5);
  int lane = threadIdx.x & 31;
  int nStrips = N >> 6;                      // strips of 64 columns
  int mTile = wave / nStrips;
  int nStrip = wave % nStrips;
  if (mTile >= (M >> 5)) return;
  int Mbase = mTile << 5;
  int Nbase = nStrip << 6;
  int l = lane & 15;
  int half = lane >> 4;

  const _Float16* a0h = Ahi + (size_t)(Mbase + l) * K + half * 8;
  const _Float16* a0l = Alo + (size_t)(Mbase + l) * K + half * 8;
  const _Float16* a1h = a0h + (size_t)16 * K;
  const _Float16* a1l = a0l + (size_t)16 * K;
  const _Float16* bhp = Bhi + (size_t)(Nbase + l) * K + half * 8;
  const _Float16* blp = Blo + (size_t)(Nbase + l) * K + half * 8;

  v8f acc0[4] = {};
  v8f acc1[4] = {};
  for (int kb = 0; kb < K; kb += 32) {
    v16h A0h = dyson_frag(a0h + kb);
    v16h A0l = dyson_frag(a0l + kb);
    v16h A1h = dyson_frag(a1h + kb);
    v16h A1l = dyson_frag(a1l + kb);
#pragma unroll
    for (int j = 0; j < 4; ++j) {
      const size_t boff = (size_t)(j * 16) * K + kb;
      v16h Bh = dyson_frag(bhp + boff);
      v16h Bl = dyson_frag(blp + boff);
      acc0[j] = __builtin_amdgcn_wmma_f32_16x16x32_f16(
          false, A0h, false, Bh, (short)0, acc0[j], false, false);
      acc0[j] = __builtin_amdgcn_wmma_f32_16x16x32_f16(
          false, A0h, false, Bl, (short)0, acc0[j], false, false);
      acc0[j] = __builtin_amdgcn_wmma_f32_16x16x32_f16(
          false, A0l, false, Bh, (short)0, acc0[j], false, false);
      acc1[j] = __builtin_amdgcn_wmma_f32_16x16x32_f16(
          false, A1h, false, Bh, (short)0, acc1[j], false, false);
      acc1[j] = __builtin_amdgcn_wmma_f32_16x16x32_f16(
          false, A1h, false, Bl, (short)0, acc1[j], false, false);
      acc1[j] = __builtin_amdgcn_wmma_f32_16x16x32_f16(
          false, A1l, false, Bh, (short)0, acc1[j], false, false);
    }
  }
  // C/D layout: lanes 0-15 -> N=lane, M=r ; lanes 16-31 -> N=lane-16, M=r+8
  int row0 = Mbase + half * 8;
  int row1 = Mbase + 16 + half * 8;
#pragma unroll
  for (int j = 0; j < 4; ++j) {
    int col = Nbase + j * 16 + l;
#pragma unroll
    for (int r = 0; r < 8; ++r) {
      size_t idx0 = (size_t)(row0 + r) * N + col;
      float v0 = acc0[j][r];
      if (gmul) v0 *= gmul[idx0];
      out[idx0] = v0;
      size_t idx1 = (size_t)(row1 + r) * N + col;
      float v1 = acc1[j][r];
      if (gmul) v1 *= gmul[idx1];
      out[idx1] = v1;
    }
  }
}

// --------------------------------------------- diagonal SSM scan (wave32)
// One wave per (b,h). Lane owns modes n=lane and n=lane+32.
// u is streamed in 32-timestep chunks via async-to-LDS double buffering;
// counted s_wait_asynccnt releases the old chunk while the next is in flight.
__device__ __forceinline__ void dyson_async_ld_b32(unsigned lds_addr,
                                                   const float* gptr) {
  asm volatile("global_load_async_to_lds_b32 %0, %1, off"
               :: "v"(lds_addr),
                  "v"((unsigned long long)(uintptr_t)gptr)
               : "memory");
}

__global__ void dyson_scan(const float* __restrict__ u,
                           const float* __restrict__ lamr,
                           const float* __restrict__ lami,
                           const float* __restrict__ wre,
                           const float* __restrict__ wim,
                           _Float16* __restrict__ yhi,
                           _Float16* __restrict__ ylo) {
  __shared__ float ub[8][2][2][32];   // [wave][buf][re/im][t-in-chunk]
  int w = threadIdx.x >> 5;
  int lane = threadIdx.x & 31;
  int wave = blockIdx.x * 8 + w;
  if (wave >= Bn * Hn) return;
  int h = wave % Hn;
  int b = wave / Hn;
  int base = h * Nmod;

  float l0r = lamr[base + lane],      l0i = lami[base + lane];
  float l1r = lamr[base + lane + 32], l1i = lami[base + lane + 32];
  float w0r = wre[base + lane],       w0i = wim[base + lane];
  float w1r = wre[base + lane + 32],  w1i = wim[base + lane + 32];

  const float* up = u + (size_t)b * Ln * Dn + h;
  _Float16* yhp = yhi + (size_t)b * Ln * Dn + h;
  _Float16* ylp = ylo + (size_t)b * Ln * Dn + h;

  // LDS byte offset of this wave's buffer (generic ptr low 32 bits = LDS addr)
  unsigned lbase = (unsigned)(uintptr_t)(&ub[w][0][0][0]);

  // prologue: issue chunk 0 (re, im) into buf 0
  dyson_async_ld_b32(lbase + lane * 4, up + (size_t)lane * Dn);
  dyson_async_ld_b32(lbase + 128 + lane * 4, up + (size_t)lane * Dn + Hn);

  float s0r = 0.f, s0i = 0.f, s1r = 0.f, s1i = 0.f;
  constexpr int NCH = Ln / 32;        // 64 chunks
  for (int c = 0; c < NCH; ++c) {
    int buf = c & 1;
    if (c + 1 < NCH) {
      const float* nup = up + (size_t)(c + 1) * 32 * Dn;
      unsigned nb = lbase + ((c + 1) & 1) * 256;
      dyson_async_ld_b32(nb + lane * 4, nup + (size_t)lane * Dn);
      dyson_async_ld_b32(nb + 128 + lane * 4, nup + (size_t)lane * Dn + Hn);
      asm volatile("s_wait_asynccnt 0x2" ::: "memory");  // chunk c complete
    } else {
      asm volatile("s_wait_asynccnt 0x0" ::: "memory");
    }
    float kyr = 0.f, kyi = 0.f;
#pragma unroll 4
    for (int j = 0; j < 32; ++j) {
      float ur = ub[w][buf][0][j];    // LDS broadcast read
      float ui = ub[w][buf][1][j];
      // s = lambda * s + u  (complex, two modes per lane)
      float n0 = fmaf(l0r, s0r, fmaf(-l0i, s0i, ur));
      s0i = fmaf(l0r, s0i, fmaf(l0i, s0r, ui));
      s0r = n0;
      float n1 = fmaf(l1r, s1r, fmaf(-l1i, s1i, ur));
      s1i = fmaf(l1r, s1i, fmaf(l1i, s1r, ui));
      s1r = n1;
      // c = w * s, summed over this lane's two modes
      float cr = fmaf(w0r, s0r, fmaf(-w0i, s0i, fmaf(w1r, s1r, -w1i * s1i)));
      float ci = fmaf(w0r, s0i, fmaf(w0i, s0r, fmaf(w1r, s1i, w1i * s1r)));
#pragma unroll
      for (int m = 16; m >= 1; m >>= 1) {
        cr += __shfl_xor(cr, m, 32);
        ci += __shfl_xor(ci, m, 32);
      }
      // lane j keeps timestep t0+j result -> 32-wide parallel store later
      kyr = (lane == j) ? cr : kyr;
      kyi = (lane == j) ? ci : kyi;
    }
    size_t idx = (size_t)(c * 32 + lane) * Dn;
    _Float16 hr = (_Float16)kyr;
    yhp[idx] = hr;  ylp[idx] = (_Float16)(kyr - (float)hr);
    _Float16 hiv = (_Float16)kyi;
    yhp[idx + Hn] = hiv;  ylp[idx + Hn] = (_Float16)(kyi - (float)hiv);
  }
}

// ------------------------------------------------------------------ launch
extern "C" void kernel_launch(void* const* d_in, const int* in_sizes, int n_in,
                              void* d_out, int out_size, void* d_ws, size_t ws_size,
                              hipStream_t stream) {
  const float* x           = (const float*)d_in[0];
  const float* g           = (const float*)d_in[1];
  const float* w_in        = (const float*)d_in[2];
  const float* conv_kernel = (const float*)d_in[3];
  const float* conv_bias   = (const float*)d_in[4];
  const float* log_decay   = (const float*)d_in[5];
  const float* freq        = (const float*)d_in[6];
  const float* w_re        = (const float*)d_in[7];
  const float* w_im        = (const float*)d_in[8];
  const float* w_out       = (const float*)d_in[9];
  float* out = (float*)d_out;

  const size_t BLD = (size_t)Bn * Ln * Dn;          // 16,777,216
  char* ws = (char*)d_ws;
  // workspace layout (reused buffers):
  float*    gact  = (float*)ws;                          // BLD*4  (also u)
  _Float16* fhi   = (_Float16*)(ws + BLD * 4);           // BLD*2  (x hi, then y hi)
  _Float16* flo   = (_Float16*)(ws + BLD * 4 + BLD * 2); // BLD*2  (x lo, then y lo)
  char* wsp = ws + BLD * 8;
  _Float16* wiThi = (_Float16*)(wsp);                    // D*D*2
  _Float16* wiTlo = (_Float16*)(wsp + 2 * 1024 * 1024);
  _Float16* woThi = (_Float16*)(wsp + 4 * 1024 * 1024);
  _Float16* woTlo = (_Float16*)(wsp + 6 * 1024 * 1024);
  float*    lamr  = (float*)   (wsp + 8 * 1024 * 1024);  // H*N*4
  float*    lami  = (float*)   (wsp + 8 * 1024 * 1024 + Hn * Nmod * 4);

  const int TB = 256;
  // 1) gated conv branch
  dyson_conv_silu<<<(int)((BLD + TB - 1) / TB), TB, 0, stream>>>(
      g, conv_kernel, conv_bias, gact);
  // 2) split x into f16 hi/lo
  dyson_split<<<(int)((BLD + TB - 1) / TB), TB, 0, stream>>>(x, fhi, flo, BLD);
  // 3) transpose+split weights
  dyson_transpose_split<<<(1024 * 1024 + TB - 1) / TB, TB, 0, stream>>>(
      w_in, wiThi, wiTlo, Dn, Dn);
  dyson_transpose_split<<<(1024 * 1024 + TB - 1) / TB, TB, 0, stream>>>(
      w_out, woThi, woTlo, Dn, Dn);
  // 4) lambda = exp(A)
  dyson_lambda<<<(Hn * Nmod + TB - 1) / TB, TB, 0, stream>>>(
      log_decay, freq, lamr, lami, Hn * Nmod);
  // 5) u = (x @ w_in) * gact   (fused epilogue; u overwrites gact in place)
  {
    const int M = Bn * Ln, N = Dn, K = Dn;
    int waves = (M / 32) * (N / 64);
    dyson_gemm_hilo<<<waves * 32 / TB, TB, 0, stream>>>(
        fhi, flo, wiThi, wiTlo, gact, gact, M, N, K);
  }
  // 6) SSM scan -> y (f16 hi/lo, overwrites x hi/lo buffers)
  dyson_scan<<<Bn * Hn / 8, TB, 0, stream>>>(
      gact, lamr, lami, w_re, w_im, fhi, flo);
  // 7) out = y @ w_out
  {
    const int M = Bn * Ln, N = Dn, K = Dn;
    int waves = (M / 32) * (N / 64);
    dyson_gemm_hilo<<<waves * 32 / TB, TB, 0, stream>>>(
        fhi, flo, woThi, woTlo, nullptr, out, M, N, K);
  }
}
